// InteractionBlock_62646392979552
// MI455X (gfx1250) — matrix operand order repaired and checked by reading, not compiled, not run on hardware
//
#include <hip/hip_runtime.h>
#include <hip/hip_bf16.h>

typedef __attribute__((ext_vector_type(16))) __bf16 v16bf;
typedef __attribute__((ext_vector_type(8)))  __bf16 v8bf;
typedef __attribute__((ext_vector_type(8)))  float  v8f;

#define E_EDGES 262144
#define W_TRIP  1048576
#define DDIM 128
#define NRBF 6
#define ADIM 49
#define JBIL 8

__device__ __forceinline__ float siluf(float x) { return x / (1.0f + __expf(-x)); }

// A fragment (16x32 bf16): lane m = lane%16; elems 0..7 -> K = k0+8*hi+e,
// elems 8..15 -> K = k0+16+8*hi+(e-8). Two contiguous 16B LDS loads.
__device__ __forceinline__ v16bf load_a_frag(const __bf16* __restrict__ act,
                                             int m, int k0, int hi) {
  union { v16bf v; v8bf h[2]; } u;
  u.h[0] = *(const v8bf*)(act + m * DDIM + k0 + 8 * hi);
  u.h[1] = *(const v8bf*)(act + m * DDIM + k0 + 16 + 8 * hi);
  return u.v;
}

// B fragment (32x16 bf16) from transposed weight wT[N][K]:
// lane col n = lane%16; elems 0..7 -> K = k0+8*hi+e, elems 8..15 -> K = k0+16+8*hi+(e-8).
__device__ __forceinline__ v16bf load_b_frag(const __bf16* __restrict__ wT,
                                             int n, int k0, int hi) {
  union { v16bf v; v8bf h[2]; } u;
  u.h[0] = *(const v8bf*)(wT + n * DDIM + k0 + 8 * hi);
  u.h[1] = *(const v8bf*)(wT + n * DDIM + k0 + 16 + 8 * hi);
  return u.v;
}

// 16x128 band GEMM, K=128: 4 k-steps x 8 n-tiles = 32 WMMA per wave.
__device__ __forceinline__ void band_gemm(const __bf16* __restrict__ act,
                                          const __bf16* __restrict__ wT,
                                          int wv, int lanelo, int hi, v8f acc[8]) {
#pragma unroll
  for (int kt = 0; kt < 4; ++kt) {
    v16bf a = load_a_frag(act + wv * 16 * DDIM, lanelo, kt * 32, hi);
#pragma unroll
    for (int nt = 0; nt < 8; ++nt) {
      v16bf b = load_b_frag(wT, nt * 16 + lanelo, kt * 32, hi);
      acc[nt] = __builtin_amdgcn_wmma_f32_16x16x32_bf16(
          false, a, false, b, (short)0, acc[nt], false, false);
    }
  }
}

__device__ __forceinline__ void stage_16k_bf16(const __bf16* __restrict__ g,
                                               __bf16* __restrict__ s) {
  const uint4* gs = (const uint4*)g;
  uint4* ss = (uint4*)s;
  for (int i = threadIdx.x; i < 2048; i += 256) ss[i] = gs[i];
}

// ---------------- prep kernels ----------------
__global__ void k_zero(float* p, long n) {
  long i = (long)blockIdx.x * 256 + threadIdx.x;
  if (i < n) p[i] = 0.0f;
}

__global__ void k_transpose_bf16(const float* __restrict__ w, __bf16* __restrict__ o) {
  int idx = blockIdx.x * 256 + threadIdx.x;  // 16384 exact
  int k = idx >> 7, n = idx & 127;
  o[n * 128 + k] = (__bf16)w[k * 128 + n];
}

__global__ void k_cvt_bf16(const float* __restrict__ w, __bf16* __restrict__ o, int n) {
  int i = blockIdx.x * 256 + threadIdx.x;
  if (i < n) o[i] = (__bf16)w[i];
}

// ------------- triplet side: S[e,j] = segsum(a_sbf @ a_sbf_w) -------------
__global__ void k_ta_scatter(const float* __restrict__ a_sbf,
                             const float* __restrict__ a_w,
                             const int* __restrict__ kj,
                             float* __restrict__ S) {
  __shared__ float sa[32 * ADIM];
  __shared__ float sw[ADIM * JBIL];
  __shared__ int   skj[32];
  int tid = threadIdx.x;
  long base = (long)blockIdx.x * 32;
  const float* ga = a_sbf + base * ADIM;  // 32 rows contiguous
  for (int i = tid; i < 32 * ADIM; i += 256) sa[i] = ga[i];
  for (int i = tid; i < ADIM * JBIL; i += 256) sw[i] = a_w[i];
  if (tid < 32) skj[tid] = kj[base + tid];
  __syncthreads();
  int row = tid >> 3, j = tid & 7;
  float acc = 0.0f;
#pragma unroll
  for (int r = 0; r < ADIM; ++r) acc += sa[row * ADIM + r] * sw[r * JBIL + j];
  unsafeAtomicAdd(S + (long)skj[row] * JBIL + j, acc);  // global_atomic_add_f32
}

// ------------- edge precompute: me_edge (bf16), transf (f32) -------------
__global__ void k_edge_pre(const float* __restrict__ m_ji,
                           const float* __restrict__ e_rbf,
                           const __bf16* __restrict__ nbrT,
                           const __bf16* __restrict__ mjiT,
                           const float* __restrict__ nbr_b,
                           const float* __restrict__ mji_b,
                           const float* __restrict__ e_w,
                           __bf16* __restrict__ me_g,
                           float* __restrict__ transf_g) {
  extern __shared__ char smem[];
  __bf16* sA  = (__bf16*)smem;            // 32KB act (bf16)
  __bf16* sW  = (__bf16*)(smem + 32768);  // 32KB weight
  float*  sE  = (float*)(smem + 65536);           // 128x6 rbf
  float*  sEW = (float*)(smem + 65536 + 3072);    // 6x128 rbf weight
  int tid = threadIdx.x;
  long row0 = (long)blockIdx.x * 128;

  const float* gm = m_ji + row0 * DDIM;
  for (int i = tid; i < 128 * DDIM; i += 256) sA[i] = (__bf16)gm[i];
  const float* ge = e_rbf + row0 * NRBF;
  for (int i = tid; i < 128 * NRBF; i += 256) sE[i] = ge[i];
  for (int i = tid; i < NRBF * DDIM; i += 256) sEW[i] = e_w[i];
  stage_16k_bf16(nbrT, sW);
  __syncthreads();

  int lane = tid & 31, lanelo = lane & 15, hi = lane >> 4, wv = tid >> 5;

  v8f acc[8];
#pragma unroll
  for (int nt = 0; nt < 8; ++nt)
#pragma unroll
    for (int r = 0; r < 8; ++r) acc[nt][r] = 0.0f;
  band_gemm(sA, sW, wv, lanelo, hi, acc);

#pragma unroll
  for (int nt = 0; nt < 8; ++nt) {
    int n = nt * 16 + lanelo;
    float b = nbr_b[n];
#pragma unroll
    for (int r = 0; r < 8; ++r) {
      int ml = wv * 16 + hi * 8 + r;
      float te = 0.0f;
#pragma unroll
      for (int q = 0; q < NRBF; ++q) te += sE[ml * NRBF + q] * sEW[q * DDIM + n];
      float v = siluf(acc[nt][r] + b) * te;
      me_g[(row0 + ml) * DDIM + n] = (__bf16)v;
    }
  }

  __syncthreads();
  stage_16k_bf16(mjiT, sW);
  __syncthreads();

  v8f acc2[8];
#pragma unroll
  for (int nt = 0; nt < 8; ++nt)
#pragma unroll
    for (int r = 0; r < 8; ++r) acc2[nt][r] = 0.0f;
  band_gemm(sA, sW, wv, lanelo, hi, acc2);

#pragma unroll
  for (int nt = 0; nt < 8; ++nt) {
    int n = nt * 16 + lanelo;
    float b = mji_b[n];
#pragma unroll
    for (int r = 0; r < 8; ++r) {
      int ml = wv * 16 + hi * 8 + r;
      transf_g[(row0 + ml) * DDIM + n] = siluf(acc2[nt][r] + b);
    }
  }
}

// ------------- directed (K=1024 GEMM) + fused 7-layer chain -------------
__global__ void k_chain(const __bf16* __restrict__ me_g,
                        const float* __restrict__ transf_g,
                        const float* __restrict__ S,
                        const __bf16* __restrict__ fT,   // [128 n][1024 k] bf16
                        const __bf16* __restrict__ wT,   // 9 x [128][128] bf16
                        const float* __restrict__ post_b,
                        const float* __restrict__ res_b,
                        float* __restrict__ out) {
  extern __shared__ char smem[];
  __bf16* sME   = (__bf16*)smem;             // 32KB me tile
  __bf16* sACT0 = (__bf16*)(smem + 32768);   // 32KB act buf 0
  __bf16* sACT1 = (__bf16*)(smem + 65536);   // 32KB act buf 1
  __bf16* sW    = (__bf16*)(smem + 98304);   // 32KB weight
  float*  sS    = (float*)(smem + 131072);   // 4KB S tile
  int tid = threadIdx.x;
  long row0 = (long)blockIdx.x * 128;

  { const uint4* g = (const uint4*)(me_g + row0 * DDIM);
    uint4* s = (uint4*)sME;
    for (int i = tid; i < 2048; i += 256) s[i] = g[i]; }
  { const float* g = S + row0 * JBIL;
    for (int i = tid; i < 128 * JBIL; i += 256) sS[i] = g[i]; }

  int lane = tid & 31, lanelo = lane & 15, hi = lane >> 4, wv = tid >> 5;

  // ---- directed = A2 @ Wf : loop j, A2 = S[:,j] (x) me ----
  v8f acc[8];
#pragma unroll
  for (int nt = 0; nt < 8; ++nt)
#pragma unroll
    for (int r = 0; r < 8; ++r) acc[nt][r] = 0.0f;

  for (int j = 0; j < JBIL; ++j) {
    __syncthreads();
    { const uint4* g = (const uint4*)fT;   // 128 uint4 per n-row
      uint4* s = (uint4*)sW;
      for (int i = tid; i < 2048; i += 256) {
        int n = i >> 4, q = i & 15;
        s[i] = g[n * 128 + j * 16 + q];
      } }
    __syncthreads();
    float sj = sS[(wv * 16 + lanelo) * JBIL + j];
#pragma unroll
    for (int kt = 0; kt < 4; ++kt) {
      v16bf araw = load_a_frag(sME + wv * 16 * DDIM, lanelo, kt * 32, hi);
      v16bf a;
#pragma unroll
      for (int e = 0; e < 16; ++e) a[e] = (__bf16)((float)araw[e] * sj);
#pragma unroll
      for (int nt = 0; nt < 8; ++nt) {
        v16bf b = load_b_frag(sW, nt * 16 + lanelo, kt * 32, hi);
        acc[nt] = __builtin_amdgcn_wmma_f32_16x16x32_bf16(
            false, a, false, b, (short)0, acc[nt], false, false);
      }
    }
  }

  // x0 = directed + transf ; keep skip (sk) and transf (tf) in f32 regs
  float tf[8][8], sk[8][8];
#pragma unroll
  for (int nt = 0; nt < 8; ++nt) {
    int n = nt * 16 + lanelo;
#pragma unroll
    for (int r = 0; r < 8; ++r) {
      int ml = wv * 16 + hi * 8 + r;
      float t = transf_g[(row0 + ml) * DDIM + n];
      tf[nt][r] = t;
      float x = acc[nt][r] + t;
      sk[nt][r] = x;
      sACT0[ml * DDIM + n] = (__bf16)x;
    }
  }

  // chain layers: res00, res01, post, res10, res11, res20, res21
  const __bf16* cw[7] = { wT + 2 * 16384, wT + 3 * 16384, wT + 4 * 16384,
                          wT + 5 * 16384, wT + 6 * 16384, wT + 7 * 16384,
                          wT + 8 * 16384 };
  const float* cb[7] = { res_b + 0, res_b + 128, post_b,
                         res_b + 256, res_b + 384, res_b + 512, res_b + 640 };
  const int ltype[7] = { 0, 1, 2, 0, 1, 0, 3 };  // 0=plain 1=+sk 2=+tf 3=+sk,out

  int cur = 0;
#pragma unroll
  for (int L = 0; L < 7; ++L) {
    __syncthreads();
    stage_16k_bf16(cw[L], sW);
    __syncthreads();
    const __bf16* actin = cur ? sACT1 : sACT0;
    __bf16* actout = cur ? sACT0 : sACT1;
    v8f a2[8];
#pragma unroll
    for (int nt = 0; nt < 8; ++nt)
#pragma unroll
      for (int r = 0; r < 8; ++r) a2[nt][r] = 0.0f;
    band_gemm(actin, sW, wv, lanelo, hi, a2);
#pragma unroll
    for (int nt = 0; nt < 8; ++nt) {
      int n = nt * 16 + lanelo;
      float b = cb[L][n];
#pragma unroll
      for (int r = 0; r < 8; ++r) {
        int ml = wv * 16 + hi * 8 + r;
        float y = siluf(a2[nt][r] + b);
        if (ltype[L] == 1)      { y += sk[nt][r]; sk[nt][r] = y; }
        else if (ltype[L] == 2) { y += tf[nt][r]; sk[nt][r] = y; }
        else if (ltype[L] == 3) { y += sk[nt][r]; }
        if (ltype[L] == 3) out[(row0 + ml) * DDIM + n] = y;
        else               actout[ml * DDIM + n] = (__bf16)y;
      }
    }
    cur ^= 1;
  }
}

extern "C" void kernel_launch(void* const* d_in, const int* in_sizes, int n_in,
                              void* d_out, int out_size, void* d_ws, size_t ws_size,
                              hipStream_t stream) {
  const float* m_ji    = (const float*)d_in[0];
  const float* e_rbf   = (const float*)d_in[1];
  const float* a_sbf   = (const float*)d_in[2];
  const int*   kj_idx  = (const int*)d_in[5];
  const float* nbr_m_w = (const float*)d_in[6];
  const float* nbr_m_b = (const float*)d_in[7];
  const float* e_rbf_w = (const float*)d_in[8];
  const float* a_sbf_w = (const float*)d_in[9];
  const float* final_w = (const float*)d_in[10];
  const float* m_ji_w  = (const float*)d_in[11];
  const float* m_ji_b  = (const float*)d_in[12];
  const float* post_w  = (const float*)d_in[13];
  const float* post_b  = (const float*)d_in[14];
  const float* res_w   = (const float*)d_in[15];
  const float* res_b   = (const float*)d_in[16];

  char* ws = (char*)d_ws;
  float*  S      = (float*)ws;                                   // 8 MB
  __bf16* me     = (__bf16*)(ws + 8388608);                      // 64 MB
  float*  transf = (float*)(ws + 8388608 + 67108864);            // 128 MB
  __bf16* wT     = (__bf16*)(ws + 8388608 + 67108864 + 134217728);        // 9*32 KB
  __bf16* fT     = (__bf16*)(ws + 8388608 + 67108864 + 134217728 + 294912); // 256 KB

  k_zero<<<(E_EDGES * 8 + 255) / 256, 256, 0, stream>>>(S, (long)E_EDGES * 8);

  k_transpose_bf16<<<64, 256, 0, stream>>>(nbr_m_w,           wT + 0 * 16384);
  k_transpose_bf16<<<64, 256, 0, stream>>>(m_ji_w,            wT + 1 * 16384);
  k_transpose_bf16<<<64, 256, 0, stream>>>(res_w + 0 * 16384, wT + 2 * 16384);
  k_transpose_bf16<<<64, 256, 0, stream>>>(res_w + 1 * 16384, wT + 3 * 16384);
  k_transpose_bf16<<<64, 256, 0, stream>>>(post_w,            wT + 4 * 16384);
  k_transpose_bf16<<<64, 256, 0, stream>>>(res_w + 2 * 16384, wT + 5 * 16384);
  k_transpose_bf16<<<64, 256, 0, stream>>>(res_w + 3 * 16384, wT + 6 * 16384);
  k_transpose_bf16<<<64, 256, 0, stream>>>(res_w + 4 * 16384, wT + 7 * 16384);
  k_transpose_bf16<<<64, 256, 0, stream>>>(res_w + 5 * 16384, wT + 8 * 16384);
  // final_w [i][j][l] is already [N=i][K=j*128+l]: convert only.
  k_cvt_bf16<<<(131072 + 255) / 256, 256, 0, stream>>>(final_w, fT, 131072);

  k_ta_scatter<<<W_TRIP / 32, 256, 0, stream>>>(a_sbf, a_sbf_w, kj_idx, S);

  size_t smemB = 32768 + 32768 + 3072 + 3072;
  k_edge_pre<<<E_EDGES / 128, 256, smemB, stream>>>(
      m_ji, e_rbf, wT + 0 * 16384, wT + 1 * 16384, nbr_m_b, m_ji_b, e_rbf_w,
      me, transf);

  size_t smemC = 4 * 32768 + 4096;
  k_chain<<<E_EDGES / 128, 256, smemC, stream>>>(
      me, transf, S, fT, wT, post_b, res_b, (float*)d_out);
}
